// QuantumModelv2_43628277793545
// MI455X (gfx1250) — matrix operand order repaired and checked by reading, not compiled, not run on hardware
//
#include <hip/hip_runtime.h>

// ---------------------------------------------------------------------------
// Types for CDNA5 WMMA (wave32)
// ---------------------------------------------------------------------------
typedef __attribute__((ext_vector_type(16))) __bf16          v16bf;
typedef __attribute__((ext_vector_type(2)))  __bf16          v2bf;
typedef __attribute__((ext_vector_type(2)))  float           v2f;
typedef __attribute__((ext_vector_type(8)))  float           v8f;
typedef __attribute__((ext_vector_type(8)))  unsigned int    v8u;

// fp32 -> bf16 via native fptrunc (RNE; backend selects v_cvt_*bf16* when present)
__device__ __forceinline__ unsigned short f2bf(float f) {
    __bf16 h = (__bf16)f;
    return __builtin_bit_cast(unsigned short, h);
}
// packed fp32x2 -> bf16x2 (vector fptrunc -> v_cvt_pk_bf16_f32 when available)
__device__ __forceinline__ unsigned int pack2bf(float a, float b) {
    v2f f; f[0] = a; f[1] = b;
    v2bf h = __builtin_convertvector(f, v2bf);
    return __builtin_bit_cast(unsigned int, h);
}
__device__ __forceinline__ float bf2f(unsigned short u) {
    return __uint_as_float(((unsigned int)u) << 16);
}

__device__ __forceinline__ float mishf(float x) {
    float sp = (x > 15.0f) ? x : __logf(1.0f + __expf(x));
    return x * tanhf(sp);
}

// two aligned 16B LDS chunks -> one WMMA 16-bit operand fragment
__device__ __forceinline__ v8u ld_frag_u(const unsigned short* p0, const unsigned short* p1) {
    uint4 lo = *(const uint4*)p0;
    uint4 hi = *(const uint4*)p1;
    v8u u;
    u[0] = lo.x; u[1] = lo.y; u[2] = lo.z; u[3] = lo.w;
    u[4] = hi.x; u[5] = hi.y; u[6] = hi.z; u[7] = hi.w;
    return u;
}
__device__ __forceinline__ v16bf as_bf(v8u u) { return __builtin_bit_cast(v16bf, u); }

// pack a D-tile column (8 consecutive rows per lane) into 16 bytes
__device__ __forceinline__ uint4 pack8(const v8f& c) {
    uint4 r;
    r.x = pack2bf(c[0], c[1]);
    r.y = pack2bf(c[2], c[3]);
    r.z = pack2bf(c[4], c[5]);
    r.w = pack2bf(c[6], c[7]);
    return r;
}

// ---------------------------------------------------------------------------
// Kernel 1: complex 32x32 recurrence per batch -> feat (B x 2048) bf16.
// LDS holds bf16 in fragment-native layouts:
//   A operands row-major [row][k]   (frag = 2x ds_load_b128)
//   B operands transposed [col][k]  (frag = 2x ds_load_b128)
//   D store: per lane col fixed, rows contiguous -> 1x ds_store_b128
// 128 threads (4 waves); each wave owns one 16x16 tile; K=32 in one WMMA.
// ---------------------------------------------------------------------------

// stage a 32x32 fp32 matrix pair row-major into bf16 LDS [32][32]
__device__ __forceinline__ void stage_rm(unsigned short* Sr, unsigned short* Si,
                                         const float* __restrict__ gr,
                                         const float* __restrict__ gi, int t) {
    int n = t * 8;                                  // 128 threads x 8 floats
    float4 a0 = *(const float4*)(gr + n);
    float4 a1 = *(const float4*)(gr + n + 4);
    float4 b0 = *(const float4*)(gi + n);
    float4 b1 = *(const float4*)(gi + n + 4);
    uint4 pr, pi;
    pr.x = pack2bf(a0.x, a0.y); pr.y = pack2bf(a0.z, a0.w);
    pr.z = pack2bf(a1.x, a1.y); pr.w = pack2bf(a1.z, a1.w);
    pi.x = pack2bf(b0.x, b0.y); pi.y = pack2bf(b0.z, b0.w);
    pi.z = pack2bf(b1.x, b1.y); pi.w = pack2bf(b1.z, b1.w);
    *(uint4*)(Sr + n) = pr;
    *(uint4*)(Si + n) = pi;
}

// stage a 32x32 fp32 matrix pair TRANSPOSED into bf16 LDS [col][row]
__device__ __forceinline__ void stage_tr(unsigned short* Sr, unsigned short* Si,
                                         const float* __restrict__ gr,
                                         const float* __restrict__ gi, int t) {
    int col = t >> 2;             // 0..31
    int r0  = (t & 3) * 8;        // 0,8,16,24
    uint4 pr, pi;
    float a[8], b[8];
#pragma unroll
    for (int j = 0; j < 8; ++j) {
        a[j] = gr[(r0 + j) * 32 + col];
        b[j] = gi[(r0 + j) * 32 + col];
    }
    pr.x = pack2bf(a[0], a[1]); pr.y = pack2bf(a[2], a[3]);
    pr.z = pack2bf(a[4], a[5]); pr.w = pack2bf(a[6], a[7]);
    pi.x = pack2bf(b[0], b[1]); pi.y = pack2bf(b[2], b[3]);
    pi.z = pack2bf(b[4], b[5]); pi.w = pack2bf(b[6], b[7]);
    *(uint4*)(Sr + col * 32 + r0) = pr;
    *(uint4*)(Si + col * 32 + r0) = pi;
}

// complex 16x16 tile:  Cr = Ar*Br + (-Ai)*Bi ;  Ci = Ar*Bi + Ai*Br
__device__ __forceinline__ void cwmma_bf(const unsigned short* Ar, const unsigned short* Ai,
                                         const unsigned short* Br, const unsigned short* Bi,
                                         int tr, int tc, int lane, v8f& cr, v8f& ci) {
    const int row = (tr + (lane & 15)) * 32;
    const int kb  = (lane >> 4) << 3;               // A: K 0..7 / 8..15 (+16)
    const int col = (tc + (lane & 15)) * 32;
    const int ks  = (lane >> 4) << 3;               // B: K 0..7 / 8..15 (+16)
    v8u ar = ld_frag_u(Ar + row + kb, Ar + row + 16 + kb);
    v8u ai = ld_frag_u(Ai + row + kb, Ai + row + 16 + kb);
    v8u an;
#pragma unroll
    for (int e = 0; e < 8; ++e) an[e] = ai[e] ^ 0x80008000u;
    v8u br = ld_frag_u(Br + col + ks, Br + col + 16 + ks);
    v8u bi = ld_frag_u(Bi + col + ks, Bi + col + 16 + ks);
    v8f z;
#pragma unroll
    for (int v = 0; v < 8; ++v) z[v] = 0.0f;
    cr = __builtin_amdgcn_wmma_f32_16x16x32_bf16(false, as_bf(ar), false, as_bf(br), (short)0, z,  false, false);
    cr = __builtin_amdgcn_wmma_f32_16x16x32_bf16(false, as_bf(an), false, as_bf(bi), (short)0, cr, false, false);
    ci = __builtin_amdgcn_wmma_f32_16x16x32_bf16(false, as_bf(ar), false, as_bf(bi), (short)0, z,  false, false);
    ci = __builtin_amdgcn_wmma_f32_16x16x32_bf16(false, as_bf(ai), false, as_bf(br), (short)0, ci, false, false);
}

__global__ __launch_bounds__(128) void recur_kernel(
    const float* __restrict__ xr, const float* __restrict__ xi,
    const float* __restrict__ ur, const float* __restrict__ ui,
    const float* __restrict__ Mr, const float* __restrict__ Mi,
    unsigned short* __restrict__ feat)
{
    __shared__ unsigned short aMr[32 * 32], aMi[32 * 32];   // A: Ms[l-1], row-major
    __shared__ unsigned short aXr[32 * 32], aXi[32 * 32];   // A: x[b,l],  row-major
    __shared__ unsigned short bOr[32 * 32], bOi[32 * 32];   // B: out,  [col][k]
    __shared__ unsigned short bTr[32 * 32], bTi[32 * 32];   // B: temp, [col][k]

    const int b    = blockIdx.x;
    const int t    = threadIdx.x;
    const int lane = t & 31;
    const int wave = t >> 5;
    const int tr   = (wave >> 1) << 4;    // 0 or 16
    const int tc   = (wave & 1) << 4;     // 0 or 16
    const size_t xb = (size_t)b * 8 * 1024;

    // D-tile store coordinates (transposed layout): 16B per lane, aligned
    const int dcol = tc + (lane & 15);
    const int drb  = tr + ((lane >> 4) << 3);

    stage_tr(bOr, bOi, xr + xb, xi + xb, t);        // out = x[b,0] (B layout)
    __syncthreads();

    v8f cr, ci;
    for (int l = 1; l < 8; ++l) {
        stage_rm(aMr, aMi, Mr + (size_t)(l - 1) * 1024, Mi + (size_t)(l - 1) * 1024, t);
        stage_rm(aXr, aXi, xr + xb + (size_t)l * 1024, xi + xb + (size_t)l * 1024, t);
        __syncthreads();
        // temp = M @ out
        cwmma_bf(aMr, aMi, bOr, bOi, tr, tc, lane, cr, ci);
        *(uint4*)(bTr + dcol * 32 + drb) = pack8(cr);
        *(uint4*)(bTi + dcol * 32 + drb) = pack8(ci);
        __syncthreads();
        // out = x @ temp
        cwmma_bf(aXr, aXi, bTr, bTi, tr, tc, lane, cr, ci);
        *(uint4*)(bOr + dcol * 32 + drb) = pack8(cr);
        *(uint4*)(bOi + dcol * 32 + drb) = pack8(ci);
        __syncthreads();
    }
    // final: temp = Ms[7] @ out in registers; res = u - temp -> feat bf16
    stage_rm(aMr, aMi, Mr + 7 * 1024, Mi + 7 * 1024, t);
    __syncthreads();
    cwmma_bf(aMr, aMi, bOr, bOi, tr, tc, lane, cr, ci);

    const size_t ub = (size_t)b * 1024;
#pragma unroll
    for (int v = 0; v < 8; ++v) {
        int row = drb + v;
        float fr = ur[ub + row * 32 + dcol] - cr[v];
        float fi = ui[ub + row * 32 + dcol] - ci[v];
        size_t fo = (size_t)b * 2048 + (size_t)row * 64 + dcol;
        feat[fo]      = f2bf(fr);
        feat[fo + 32] = f2bf(fi);
    }
}

// ---------------------------------------------------------------------------
// Kernel 2/3: C = mish(A(MxK bf16) @ W(KxN bf16) + bias), C bf16.
// Block 256 threads (8 waves), tile 128(M) x 64(N), K chunks of 32.
// A tile staged via GLOBAL_LOAD_ASYNC_TO_LDS_B128 (ASYNCcnt).
// ---------------------------------------------------------------------------
__global__ __launch_bounds__(256) void gemm_mish(
    const unsigned short* __restrict__ A,
    const unsigned short* __restrict__ W,
    const float* __restrict__ bias,
    unsigned short* __restrict__ C,
    int K, int N)
{
    __shared__ unsigned short As[128 * 32];   // [row][k]  8 KB
    __shared__ unsigned short Ws[64 * 32];    // [col][k]  4 KB (transposed)

    const int t    = threadIdx.x;
    const int lane = t & 31;
    const int wave = t >> 5;
    const int wm   = wave >> 1;              // 0..3
    const int wn   = wave & 1;               // 0..1
    const int rowBase = blockIdx.y * 128;
    const int colBase = blockIdx.x * 64;
    const int kb = (lane >> 4) << 3;
    const int ks = (lane >> 4) << 4;

    v8f acc[2][2];
#pragma unroll
    for (int i = 0; i < 2; ++i)
#pragma unroll
        for (int j = 0; j < 2; ++j)
#pragma unroll
            for (int v = 0; v < 8; ++v) acc[i][j][v] = 0.0f;

    for (int kc = 0; kc < K; kc += 32) {
        // --- stage A tile 128x32 asynchronously: 2 x (256 lanes * 16B = 4KB)
#pragma unroll
        for (int c = 0; c < 2; ++c) {
            int chunk = t + c * 256;                 // 0..511
            int row   = chunk >> 2;
            int seg   = (chunk & 3) * 8;
            const unsigned short* gp = A + (size_t)(rowBase + row) * K + kc + seg;
            unsigned int lds = (unsigned int)(size_t)(As + row * 32 + seg);
            asm volatile("global_load_async_to_lds_b128 %0, %1, off"
                         :: "v"(lds), "v"(gp) : "memory");
        }
        // --- stage W tile 32x64, transposed into [col][k] via VGPRs
        {
            int kr = t >> 3;                          // 0..31
            int cg = (t & 7) * 8;
            uint4 v = *(const uint4*)(W + (size_t)(kc + kr) * N + colBase + cg);
            const unsigned short* pv = (const unsigned short*)&v;
#pragma unroll
            for (int j = 0; j < 8; ++j) Ws[(cg + j) * 32 + kr] = pv[j];
        }
        if (kc + 32 < K)
            __builtin_prefetch(W + (size_t)(kc + 32 + (t & 31)) * N + colBase, 0, 1);
        asm volatile("s_wait_asynccnt 0x0" ::: "memory");
        __syncthreads();

        v16bf af[2], bfr[2];
#pragma unroll
        for (int i = 0; i < 2; ++i) {
            int row = (wm * 32 + i * 16 + (lane & 15)) * 32;
            af[i] = as_bf(ld_frag_u(As + row + kb, As + row + 16 + kb));
        }
#pragma unroll
        for (int j = 0; j < 2; ++j) {
            int col = (wn * 32 + j * 16 + (lane & 15)) * 32;
            bfr[j] = as_bf(ld_frag_u(Ws + col + kb, Ws + col + 16 + kb));
        }
#pragma unroll
        for (int i = 0; i < 2; ++i)
#pragma unroll
            for (int j = 0; j < 2; ++j)
                acc[i][j] = __builtin_amdgcn_wmma_f32_16x16x32_bf16(
                    false, af[i], false, bfr[j], (short)0, acc[i][j], false, false);
        __syncthreads();
    }

    // epilogue: bias + mish -> bf16
#pragma unroll
    for (int i = 0; i < 2; ++i)
#pragma unroll
        for (int j = 0; j < 2; ++j) {
            int col  = colBase + wn * 32 + j * 16 + (lane & 15);
            float bv = bias[col];
            int rloc = wm * 32 + i * 16 + ((lane >> 4) << 3);
#pragma unroll
            for (int v = 0; v < 8; ++v) {
                int row = rowBase + rloc + v;
                float x = acc[i][j][v] + bv;
                C[(size_t)row * N + col] = f2bf(mishf(x));
            }
        }
}

// ---------------------------------------------------------------------------
// Kernel 4: out[b] = dot(h2[b, 0:1024], W3) + b3 ; one wave per row.
// Per lane: 4 contiguous 16B bf16 chunks + 8 x float4 of W3.
// ---------------------------------------------------------------------------
__global__ __launch_bounds__(256) void dot_kernel(
    const unsigned short* __restrict__ H,
    const float* __restrict__ W3, const float* __restrict__ b3,
    float* __restrict__ out)
{
    const int lane = threadIdx.x & 31;
    const int wave = threadIdx.x >> 5;
    const int row  = blockIdx.x * 8 + wave;
    const unsigned short* h = H + (size_t)row * 1024;
    float s = 0.0f;
#pragma unroll
    for (int p = 0; p < 4; ++p) {
        int k0 = p * 256 + lane * 8;
        uint4 hv = *(const uint4*)(h + k0);
        const unsigned short* pv = (const unsigned short*)&hv;
        float4 w0 = *(const float4*)(W3 + k0);
        float4 w1 = *(const float4*)(W3 + k0 + 4);
        s += bf2f(pv[0]) * w0.x + bf2f(pv[1]) * w0.y + bf2f(pv[2]) * w0.z + bf2f(pv[3]) * w0.w;
        s += bf2f(pv[4]) * w1.x + bf2f(pv[5]) * w1.y + bf2f(pv[6]) * w1.z + bf2f(pv[7]) * w1.w;
    }
#pragma unroll
    for (int off = 16; off; off >>= 1) s += __shfl_xor(s, off);
    if (lane == 0) out[row] = s + b3[0];
}

// fp32 -> bf16 weight conversion, 8 elements per thread (n % 2048 == 0)
__global__ __launch_bounds__(256) void convbf8(const float* __restrict__ s,
                                               unsigned short* __restrict__ d, int n)
{
    int i = (blockIdx.x * 256 + threadIdx.x) * 8;
    if (i + 8 <= n) {
        float4 a = *(const float4*)(s + i);
        float4 b = *(const float4*)(s + i + 4);
        uint4 p;
        p.x = pack2bf(a.x, a.y); p.y = pack2bf(a.z, a.w);
        p.z = pack2bf(b.x, b.y); p.w = pack2bf(b.z, b.w);
        *(uint4*)(d + i) = p;
    }
}

// ---------------------------------------------------------------------------
// Launch
// ---------------------------------------------------------------------------
extern "C" void kernel_launch(void* const* d_in, const int* in_sizes, int n_in,
                              void* d_out, int out_size, void* d_ws, size_t ws_size,
                              hipStream_t stream)
{
    (void)in_sizes; (void)n_in; (void)out_size; (void)ws_size;
    const float* xr = (const float*)d_in[0];
    const float* xi = (const float*)d_in[1];
    const float* ur = (const float*)d_in[2];
    const float* ui = (const float*)d_in[3];
    /* d_in[4] = 'o' (unused by reference) */
    const float* Mr = (const float*)d_in[5];
    const float* Mi = (const float*)d_in[6];
    const float* W1 = (const float*)d_in[7];
    const float* b1 = (const float*)d_in[8];
    const float* W2 = (const float*)d_in[9];
    const float* b2 = (const float*)d_in[10];
    const float* W3 = (const float*)d_in[11];
    const float* b3 = (const float*)d_in[12];

    char* ws = (char*)d_ws;
    unsigned short* feat = (unsigned short*)(ws);                 // 8192*2048 bf16 = 32 MB
    unsigned short* h1   = (unsigned short*)(ws + 33554432u);     // 8192*1024 bf16 = 16 MB
    unsigned short* h2   = (unsigned short*)(ws + 50331648u);     // 8192*1024 bf16 = 16 MB
    unsigned short* W1b  = (unsigned short*)(ws + 67108864u);     // 2048*1024 bf16 = 4 MB
    unsigned short* W2b  = (unsigned short*)(ws + 71303168u);     // 1024*1024 bf16 = 2 MB

    convbf8<<<(2048 * 1024) / 2048, 256, 0, stream>>>(W1, W1b, 2048 * 1024);
    convbf8<<<(1024 * 1024) / 2048, 256, 0, stream>>>(W2, W2b, 1024 * 1024);

    recur_kernel<<<8192, 128, 0, stream>>>(xr, xi, ur, ui, Mr, Mi, feat);

    dim3 g1(1024 / 64, 8192 / 128);   // (N tiles, M tiles)
    gemm_mish<<<g1, 256, 0, stream>>>(feat, W1b, b1, h1, 2048, 1024);
    gemm_mish<<<g1, 256, 0, stream>>>(h1,   W2b, b2, h2, 1024, 1024);

    dot_kernel<<<8192 / 8, 256, 0, stream>>>(h2, W3, b3, (float*)d_out);
}
